// Transformer_56495999812233
// MI455X (gfx1250) — matrix operand order repaired and checked by reading, not compiled, not run on hardware
//
#include <hip/hip_runtime.h>
#include <hip/hip_bf16.h>
#include <math.h>

// ---------------------------------------------------------------------------
// Transformer (6 parallel layers, 3 full-width heads) for MI455X / gfx1250.
// All GEMMs run through one f16-WMMA kernel (v_wmma_f32_16x16x32_f16),
// f32 storage everywhere, f32->f16 conversion while staging into LDS.
// 128x128 block tile, 256 threads (8 waves), software-pipelined global loads.
// ---------------------------------------------------------------------------

typedef __attribute__((ext_vector_type(16))) _Float16 v16h;
typedef __attribute__((ext_vector_type(8)))  float    v8f;

#define LDSROW 40   // 32 K-elements + 8 f16 pad; row stride 80B (16B aligned)

union Frag16 { uint4 q[2]; v16h h; };

// C[M,N] = A[M,K] * B + bias.   b_transposed==0: B is [K,N] row-major.
//                               b_transposed==1: B is [N,K] row-major (B^T).
// Block tile 128x128, 256 threads = 8 waves in a 4(m) x 2(n) wave grid;
// each wave computes 32x64 as 2x4 WMMA 16x16 tiles, K stepped by 32.
__global__ __launch_bounds__(256)
void gemm_wmma_f16(const float* __restrict__ A, const float* __restrict__ B,
                   const float* __restrict__ bias, float* __restrict__ C,
                   int M, int N, int Kdim, int lda, int ldb, int ldc,
                   int b_transposed)
{
    __shared__ _Float16 lsA[128 * LDSROW];
    __shared__ _Float16 lsB[128 * LDSROW];

    const int tid  = threadIdx.x;
    const int lane = tid & 31;
    const int wave = tid >> 5;
    const int hi   = lane >> 4;     // which 16-lane half
    const int l15  = lane & 15;

    const int n0 = blockIdx.x * 128;
    const int m0 = blockIdx.y * 128;
    const int wm = (wave >> 1) * 32;    // wave m offset in tile (0,32,64,96)
    const int wn = (wave & 1) * 64;     // wave n offset in tile (0,64)

    v8f acc[2][4] = {};

    float4 ra[4], rb[4];

    // ---- global tile loads into registers (issued back-to-back) ---------
    auto loadA = [&](int kt) {
        #pragma unroll
        for (int it = 0; it < 4; ++it) {
            int q   = it * 256 + tid;          // 0..1023 over [128 rows][8 f4]
            int row = q >> 3;
            int col = (q & 7) << 2;
            ra[it] = *(const float4*)(A + (size_t)(m0 + row) * lda + (kt + col));
        }
    };
    auto loadB = [&](int kt) {
        if (b_transposed) {
            #pragma unroll
            for (int it = 0; it < 4; ++it) {
                int q   = it * 256 + tid;      // [128 n-rows][8 f4 of K]
                int row = q >> 3;
                int col = (q & 7) << 2;
                rb[it] = *(const float4*)(B + (size_t)(n0 + row) * ldb + (kt + col));
            }
        } else {
            #pragma unroll
            for (int it = 0; it < 4; ++it) {
                int q  = it * 256 + tid;       // [32 k][32 n4]
                int kk = q >> 5;
                int nn = (q & 31) << 2;
                rb[it] = *(const float4*)(B + (size_t)(kt + kk) * ldb + (n0 + nn));
            }
        }
    };
    // ---- register -> LDS (f32 -> f16 convert) ---------------------------
    auto storeA = [&]() {
        #pragma unroll
        for (int it = 0; it < 4; ++it) {
            int q   = it * 256 + tid;
            int row = q >> 3;
            int col = (q & 7) << 2;
            _Float16* d = &lsA[row * LDSROW + col];
            d[0] = (_Float16)ra[it].x; d[1] = (_Float16)ra[it].y;
            d[2] = (_Float16)ra[it].z; d[3] = (_Float16)ra[it].w;
        }
    };
    auto storeB = [&]() {
        if (b_transposed) {
            #pragma unroll
            for (int it = 0; it < 4; ++it) {
                int q   = it * 256 + tid;
                int row = q >> 3;
                int col = (q & 7) << 2;
                _Float16* d = &lsB[row * LDSROW + col];
                d[0] = (_Float16)rb[it].x; d[1] = (_Float16)rb[it].y;
                d[2] = (_Float16)rb[it].z; d[3] = (_Float16)rb[it].w;
            }
        } else {
            #pragma unroll
            for (int it = 0; it < 4; ++it) {
                int q  = it * 256 + tid;
                int kk = q >> 5;
                int nn = (q & 31) << 2;
                lsB[(nn + 0) * LDSROW + kk] = (_Float16)rb[it].x;
                lsB[(nn + 1) * LDSROW + kk] = (_Float16)rb[it].y;
                lsB[(nn + 2) * LDSROW + kk] = (_Float16)rb[it].z;
                lsB[(nn + 3) * LDSROW + kk] = (_Float16)rb[it].w;
            }
        }
    };

    loadA(0);
    loadB(0);

    for (int kt = 0; kt < Kdim; kt += 32) {
        __syncthreads();            // previous compute done reading LDS
        storeA();
        storeB();
        __syncthreads();

        // prefetch tile kt+64 (speculative, gfx1250 global_prefetch path)
        if (kt + 64 < Kdim) {
            __builtin_prefetch(A + (size_t)(m0 + (tid >> 1)) * lda + kt + 64, 0, 3);
            __builtin_prefetch(B + (b_transposed
                                    ? (size_t)(n0 + (tid >> 1)) * ldb + kt + 64
                                    : (size_t)(kt + 64 + (tid >> 3)) * ldb + n0), 0, 3);
        }
        // software pipeline: issue next tile's global loads now,
        // overlap them with this tile's WMMAs
        if (kt + 32 < Kdim) {
            loadA(kt + 32);
            loadB(kt + 32);
        }

        // ---- fragments per ISA 7.12.2 layouts ---------------------------
        // A 16x32: lane(0-15): elems 0..7 = K0..7, 8..15 = K16..23
        //          lane(16-31): elems 0..7 = K8..15, 8..15 = K24..31
        Frag16 af[2];
        #pragma unroll
        for (int i = 0; i < 2; ++i) {
            int row = wm + i * 16 + l15;
            af[i].q[0] = *(const uint4*)&lsA[row * LDSROW + hi * 8];
            af[i].q[1] = *(const uint4*)&lsA[row * LDSROW + 16 + hi * 8];
        }
        // B 32x16: same K order as A (elems 0..7 = K8h..8h+7, 8..15 = K16+8h..), N = lane&15
        #pragma unroll
        for (int j = 0; j < 4; ++j) {
            Frag16 bf;
            int row = wn + j * 16 + l15;
            bf.q[0] = *(const uint4*)&lsB[row * LDSROW + hi * 8];
            bf.q[1] = *(const uint4*)&lsB[row * LDSROW + 16 + hi * 8];
            #pragma unroll
            for (int i = 0; i < 2; ++i)
                acc[i][j] = __builtin_amdgcn_wmma_f32_16x16x32_f16(
                    false, af[i].h, false, bf.h,
                    (short)0, acc[i][j], false, false);
        }
    }

    // ---- write C: VGPR r holds M = r + 8*hi, N = lane&15 ----------------
    #pragma unroll
    for (int i = 0; i < 2; ++i) {
        int mrow = m0 + wm + i * 16 + hi * 8;
        #pragma unroll
        for (int j = 0; j < 4; ++j) {
            int ncol = n0 + wn + j * 16 + l15;
            float bv = bias ? bias[ncol] : 0.0f;
            #pragma unroll
            for (int r = 0; r < 8; ++r)
                C[(size_t)(mrow + r) * ldc + ncol] = acc[i][j][r] + bv;
        }
    }
    (void)M; (void)N;
}

// x[r][e] = emb_w[inputs[r]][e] + pvec[e], pvec per reference (pos = S-1)
__global__ void embed_kernel(const int* __restrict__ inputs,
                             const float* __restrict__ emb_w,
                             float* __restrict__ x)
{
    int idx = blockIdx.x * blockDim.x + threadIdx.x;   // over 2048*512
    int r = idx >> 9;
    int e = idx & 511;
    int tok = inputs[r];
    float denom = powf(10000.0f, 2.0f * (float)e / 512.0f);
    float arg   = 1023.0f / denom;
    float p = (e & 1) ? cosf(arg) : sinf(arg);
    x[idx] = emb_w[(size_t)tok * 512 + e] + p;
}

// softmax over the QUERY axis (reference axis=2) with causal-past mask:
// column t is normalized over rows s <= t; rows s > t get 0.
__global__ void softmax_col_kernel(float* __restrict__ att, int S, float scale)
{
    int t = blockIdx.x * blockDim.x + threadIdx.x;
    float* m = att + (size_t)blockIdx.y * S * S;
    if (t >= S) return;
    float mx = -3.0e38f;
    for (int s = 0; s <= t; ++s)
        mx = fmaxf(mx, m[(size_t)s * S + t] * scale);
    float sum = 0.0f;
    for (int s = 0; s <= t; ++s)
        sum += expf(m[(size_t)s * S + t] * scale - mx);
    float inv = 1.0f / sum;
    for (int s = 0; s <= t; ++s)
        m[(size_t)s * S + t] = expf(m[(size_t)s * S + t] * scale - mx) * inv;
    for (int s = t + 1; s < S; ++s)
        m[(size_t)s * S + t] = 0.0f;
}

// out[row*ldo + off + e] = LN(y + resid) * g + b   (E = 512, block = 256)
__global__ __launch_bounds__(256)
void ln_kernel(const float* __restrict__ y, const float* __restrict__ resid,
               const float* __restrict__ g, const float* __restrict__ b,
               float* __restrict__ out, int ldo, int off)
{
    __shared__ float s1[256];
    __shared__ float s2[256];
    int r   = blockIdx.x;
    int tid = threadIdx.x;
    float v0 = y[(size_t)r * 512 + tid]       + resid[(size_t)r * 512 + tid];
    float v1 = y[(size_t)r * 512 + 256 + tid] + resid[(size_t)r * 512 + 256 + tid];
    s1[tid] = v0 + v1;
    s2[tid] = v0 * v0 + v1 * v1;
    __syncthreads();
    for (int st = 128; st > 0; st >>= 1) {
        if (tid < st) { s1[tid] += s1[tid + st]; s2[tid] += s2[tid + st]; }
        __syncthreads();
    }
    float mean = s1[0] * (1.0f / 512.0f);
    float var  = s2[0] * (1.0f / 512.0f) - mean * mean;
    float rinv = rsqrtf(var + 1e-5f);
    out[(size_t)r * ldo + off + tid]       = (v0 - mean) * rinv * g[tid]       + b[tid];
    out[(size_t)r * ldo + off + 256 + tid] = (v1 - mean) * rinv * g[256 + tid] + b[256 + tid];
}

extern "C" void kernel_launch(void* const* d_in, const int* in_sizes, int n_in,
                              void* d_out, int out_size, void* d_ws, size_t ws_size,
                              hipStream_t stream)
{
    (void)in_sizes; (void)n_in; (void)out_size; (void)ws_size;
    constexpr int Bq = 2, S = 1024, E = 512, L = 6, H = 3, V = 32000;
    constexpr int R = Bq * S;               // 2048 token rows

    const int*   inputs = (const int*)  d_in[0];
    const float* emb_w  = (const float*)d_in[1];
    const float* Wq     = (const float*)d_in[2];
    const float* Wk     = (const float*)d_in[3];
    const float* Wv     = (const float*)d_in[4];
    const float* mh_w   = (const float*)d_in[5];
    const float* mh_b   = (const float*)d_in[6];
    const float* l1_w   = (const float*)d_in[7];
    const float* l1_b   = (const float*)d_in[8];
    const float* l2_w   = (const float*)d_in[9];
    const float* l2_b   = (const float*)d_in[10];
    const float* l3_w   = (const float*)d_in[11];
    const float* l3_b   = (const float*)d_in[12];
    const float* ln_g   = (const float*)d_in[13];
    const float* ln_b   = (const float*)d_in[14];
    const float* out_w  = (const float*)d_in[15];
    const float* out_b  = (const float*)d_in[16];
    float* out = (float*)d_out;

    float* ws = (float*)d_ws;
    size_t o = 0;
    float* x    = ws + o; o += (size_t)R * E;          // token embeddings
    float* Qm   = ws + o; o += (size_t)H * R * E;      // Q per head
    float* Km   = ws + o; o += (size_t)H * R * E;      // K per head
    float* Vm   = ws + o; o += (size_t)H * R * E;      // V per head
    float* att  = ws + o; o += (size_t)H * Bq * S * S; // attention scores/probs
    float* cat  = ws + o; o += (size_t)R * H * E;      // head concat
    float* amat = ws + o; o += (size_t)R * E;          // post-attn LN
    float* t1   = ws + o; o += (size_t)R * E;
    float* t2   = ws + o; o += (size_t)R * E;
    float* lin  = ws + o; o += (size_t)R * E;
    float* cat6 = ws + o; o += (size_t)R * L * E;      // layer concat (~116 MB total)

    auto gemm = [&](const float* A, const float* Bm, const float* bias, float* C,
                    int M, int N, int K, int lda, int ldb, int ldc, int bt) {
        dim3 grid(N / 128, M / 128);
        gemm_wmma_f16<<<grid, 256, 0, stream>>>(A, Bm, bias, C, M, N, K, lda, ldb, ldc, bt);
    };

    embed_kernel<<<(R * E) / 256, 256, 0, stream>>>(inputs, emb_w, x);

    const float scale = 1.0f / sqrtf((float)E);

    for (int l = 0; l < L; ++l) {
        // Q/K/V projections: [2048,512] @ [512,512] per head
        for (int h = 0; h < H; ++h) {
            size_t woff = (size_t)(l * H + h) * E * E;
            gemm(x, Wq + woff, nullptr, Qm + (size_t)h * R * E, R, E, E, E, E, E, 0);
            gemm(x, Wk + woff, nullptr, Km + (size_t)h * R * E, R, E, E, E, E, E, 0);
            gemm(x, Wv + woff, nullptr, Vm + (size_t)h * R * E, R, E, E, E, E, E, 0);
        }
        // scores = Q @ K^T  per (head, batch)  [1024,1024]
        for (int h = 0; h < H; ++h)
            for (int b = 0; b < Bq; ++b) {
                const float* qh = Qm + (size_t)h * R * E + (size_t)b * S * E;
                const float* kh = Km + (size_t)h * R * E + (size_t)b * S * E;
                float* am = att + (size_t)(h * Bq + b) * S * S;
                gemm(qh, kh, nullptr, am, S, S, E, E, E, S, 1);
            }
        softmax_col_kernel<<<dim3(S / 256, H * Bq), 256, 0, stream>>>(att, S, scale);
        // o = probs @ V, written into the head-concat layout [b*S+s][h*E+f]
        for (int h = 0; h < H; ++h)
            for (int b = 0; b < Bq; ++b) {
                const float* am = att + (size_t)(h * Bq + b) * S * S;
                const float* vh = Vm + (size_t)h * R * E + (size_t)b * S * E;
                float* co = cat + (size_t)b * S * (H * E) + (size_t)h * E;
                gemm(am, vh, nullptr, co, S, E, S, S, E, H * E, 0);
            }
        // a = LN(x + cat @ mh_w + mh_b)
        gemm(cat, mh_w + (size_t)l * (H * E) * E, mh_b + (size_t)l * E, t1,
             R, E, H * E, H * E, E, E, 0);
        ln_kernel<<<R, 256, 0, stream>>>(t1, x, ln_g + (size_t)l * E, ln_b + (size_t)l * E,
                                         amat, E, 0);
        // lin = ((a@w1+b1)@w2+b2)@w3+b3  (no nonlinearity, as in source)
        gemm(amat, l1_w + (size_t)l * E * E, l1_b + (size_t)l * E, t1, R, E, E, E, E, E, 0);
        gemm(t1,   l2_w + (size_t)l * E * E, l2_b + (size_t)l * E, t2, R, E, E, E, E, E, 0);
        gemm(t2,   l3_w + (size_t)l * E * E, l3_b + (size_t)l * E, lin, R, E, E, E, E, E, 0);
        // layer out = LN(lin + a), stored into cat6 column block l*E..
        ln_kernel<<<R, 256, 0, stream>>>(lin, amat, ln_g + (size_t)l * E, ln_b + (size_t)l * E,
                                         cat6, L * E, l * E);
    }
    // logits = cat6 @ out_w + out_b : [2048,3072]x[3072,32000]
    gemm(cat6, out_w, out_b, out, R, V, L * E, L * E, V, V, 0);
}
